// GCN_mamba_Net_Encoder_16853451669714
// MI455X (gfx1250) — compile-verified, matrix operand order checked by reading, not run on hardware
//
#include <hip/hip_runtime.h>
#include <hip/hip_bf16.h>

#define NN 20000
#define FIN 512
#define DD 128
#define DIi 256
#define DSs 16
#define EE 320000
#define LCH 128
#define NCHUNK ((NN + LCH - 1) / LCH)   // 157
#define NT 4                            // 16x64 output per wave: 4 WMMAs share one A fragment

typedef __attribute__((ext_vector_type(8)))  float   v8f;
typedef __attribute__((ext_vector_type(16))) __bf16  v16bf;
typedef __attribute__((ext_vector_type(8)))  __bf16  v8bf;

__device__ __forceinline__ float wave_red32(float v) {
#pragma unroll
  for (int off = 16; off; off >>= 1) v += __shfl_xor(v, off);
  return v;
}

// ---- generic bf16 WMMA GEMM: C[M,Nc] = A[M,K] * Bt[Nc,K]^T (+bias), 16x64 per wave ----
// B loads sit directly before their WMMA: the compiler 2x-unrolls the K loop and
// issues the next tile's loads under the current WMMA's XDL execution (round-2
// codegen: best observed mix, 12 wmma, clause-grouped b128 loads).
__global__ void k_gemm_bf16(const __bf16* __restrict__ A, const __bf16* __restrict__ Bt,
                            float* __restrict__ C, const float* __restrict__ bias,
                            int M, int K, int Nc) {
  int gwave = (int)((blockIdx.x * blockDim.x + threadIdx.x) >> 5);
  int lane  = threadIdx.x & 31;
  int ng    = Nc >> 6;                   // 64-column groups
  int tiles = (M >> 4) * ng;
  if (gwave >= tiles) return;            // wave-uniform: EXEC stays all-1 for WMMA
  int tm = gwave / ng;
  int tg = gwave - tm * ng;
  int l15 = lane & 15;
  int kh  = (lane >> 4) << 3;            // K-half select: 0 or 8
  const __bf16* arow = A + (size_t)(tm * 16 + l15) * K;
  const __bf16* brow[NT];
#pragma unroll
  for (int t = 0; t < NT; ++t)
    brow[t] = Bt + (size_t)(tg * 64 + t * 16 + l15) * K;
  v8f acc[NT] = {};
  for (int k0 = 0; k0 < K; k0 += 32) {
    v8bf alo = *(const v8bf*)(arow + k0 + kh);
    v8bf ahi = *(const v8bf*)(arow + k0 + 16 + kh);
    v16bf av;
#pragma unroll
    for (int i = 0; i < 8; ++i) { av[i] = alo[i]; av[i + 8] = ahi[i]; }
#pragma unroll
    for (int t = 0; t < NT; ++t) {
      v8bf blo = *(const v8bf*)(brow[t] + k0 + kh);
      v8bf bhi = *(const v8bf*)(brow[t] + k0 + 16 + kh);
      v16bf bv;
#pragma unroll
      for (int i = 0; i < 8; ++i) { bv[i] = blo[i]; bv[i + 8] = bhi[i]; }
      acc[t] = __builtin_amdgcn_wmma_f32_16x16x32_bf16(false, av, false, bv,
                                                       (short)0, acc[t], false, false);
    }
  }
  int orow = tm * 16 + ((lane >> 4) << 3);
#pragma unroll
  for (int t = 0; t < NT; ++t) {
    int ocol = tg * 64 + t * 16 + l15;
    float b = bias ? bias[ocol] : 0.f;
#pragma unroll
    for (int r = 0; r < 8; ++r)
      C[(size_t)(orow + r) * Nc + ocol] = acc[t][r] + b;
  }
}

// ---------------- weight prep ----------------
__global__ void k_cvt_bf16(const float* __restrict__ s, __bf16* __restrict__ d, long n) {
  long g = blockIdx.x * 256L + threadIdx.x;
  if (g < n) d[g] = (__bf16)s[g];
}
__global__ void k_w1t(const float* __restrict__ W1, __bf16* __restrict__ W1t) {
  int g = blockIdx.x * 256 + threadIdx.x;           // 128*512
  int k = g & 511, j = g >> 9;
  W1t[g] = (__bf16)W1[(size_t)k * DD + j];
}
__global__ void k_wcat(const float* __restrict__ Wf, const float* __restrict__ Wv1,
                       const float* __restrict__ Wv2, __bf16* __restrict__ Wc) {
  int g = blockIdx.x * 256 + threadIdx.x;           // 384*256
  int k = g & 255, j = g >> 8;
  float v;
  if (j < 128)        v = Wf[(size_t)k * DD + j];
  else if (j < 256)   v = (k < 128) ? Wv1[(size_t)k * DD + (j - 128)] : 0.f;
  else                v = (k >= 128) ? Wv2[(size_t)(k - 128) * DD + (j - 256)] : 0.f;
  Wc[g] = (__bf16)v;
}

// ---------------- LayerNorm + Gumbel straight-through gate (1 wave / row) ----------------
__global__ void k_ln_gate(const float* __restrict__ xin, const float* __restrict__ lw,
                          const float* __restrict__ lb, const float* __restrict__ Wg,
                          const float* __restrict__ bg, const float* __restrict__ gu,
                          float* __restrict__ xnorm, __bf16* __restrict__ hcat,
                          float* __restrict__ gate) {
  int row = blockIdx.x * 8 + (threadIdx.x >> 5);
  if (row >= NN) return;
  int lane = threadIdx.x & 31;
  float4 xv = *(const float4*)(xin + (size_t)row * DD + lane * 4);
  const float* xp = (const float*)&xv;
  float s = xv.x + xv.y + xv.z + xv.w;
  float s2 = xv.x*xv.x + xv.y*xv.y + xv.z*xv.z + xv.w*xv.w;
  s = wave_red32(s); s2 = wave_red32(s2);
  float m = s * (1.f / DD);
  float inv = rsqrtf(s2 * (1.f / DD) - m * m + 1e-5f);
  float o[4], l0 = 0.f, l1 = 0.f;
#pragma unroll
  for (int t = 0; t < 4; ++t) {
    int j = lane * 4 + t;
    float xn = (xp[t] - m) * inv * lw[j] + lb[j];
    o[t] = xn;
    l0 += xn * Wg[j * 2 + 0];
    l1 += xn * Wg[j * 2 + 1];
  }
  *(float4*)(xnorm + (size_t)row * DD + lane * 4) = make_float4(o[0], o[1], o[2], o[3]);
  __bf16* hc = hcat + (size_t)row * (2 * DD) + lane * 4;
  hc[0] = (__bf16)o[0]; hc[1] = (__bf16)o[1]; hc[2] = (__bf16)o[2]; hc[3] = (__bf16)o[3];
  l0 = wave_red32(l0); l1 = wave_red32(l1);
  if (lane == 0) {
    float u0 = fminf(fmaxf(gu[row * 2 + 0], 1e-6f), 1.f - 1e-6f);
    float u1 = fminf(fmaxf(gu[row * 2 + 1], 1e-6f), 1.f - 1e-6f);
    float g0 = -logf(-logf(u0)), g1 = -logf(-logf(u1));
    gate[row] = ((l1 + bg[1] + g1) > (l0 + bg[0] + g0)) ? 1.f : 0.f;  // y_hard (straight-through)
  }
}

// ---------------- gated edge scatter-add (1 wave / edge, f32 atomics) ----------------
__global__ void k_scatter(const int* __restrict__ ei, const float* __restrict__ ew,
                          const float* __restrict__ gate, const float* __restrict__ xnorm,
                          float* __restrict__ view2) {
  int e = blockIdx.x * 8 + (threadIdx.x >> 5);
  if (e >= EE) return;
  int lane = threadIdx.x & 31;
  int row = ei[e], col = ei[EE + e];
  float w = ew[e] * gate[row] * gate[col];
  if (w == 0.f) return;                           // ~3/4 of edges die (binary gates)
  float4 v = *(const float4*)(xnorm + (size_t)col * DD + lane * 4);
  float* dst = view2 + (size_t)row * DD + lane * 4;
  atomicAdd(dst + 0, w * v.x); atomicAdd(dst + 1, w * v.y);
  atomicAdd(dst + 2, w * v.z); atomicAdd(dst + 3, w * v.w);
}

__global__ void k_v2hcat(const float* __restrict__ v2, __bf16* __restrict__ hcat) {
  long g = blockIdx.x * 256L + threadIdx.x;       // N*128
  if (g >= (long)NN * DD) return;
  int j = (int)(g & (DD - 1)); long n = g >> 7;
  hcat[n * (2 * DD) + DD + j] = (__bf16)v2[g];
}

// ---------------- gated fusion epilogue ----------------
__global__ void k_fused(const float* __restrict__ t384, const float* __restrict__ bfv,
                        const float* __restrict__ bv1, const float* __restrict__ bv2,
                        __bf16* __restrict__ fused) {
  long g = blockIdx.x * 256L + threadIdx.x;
  if (g >= (long)NN * DD) return;
  int j = (int)(g & (DD - 1)); long n = g >> 7;
  const float* tr = t384 + n * 384;
  float fg = 1.f / (1.f + __expf(-(tr[j] + bfv[j])));
  float a  = tr[128 + j] + bv1[j];
  float b  = tr[256 + j] + bv2[j];
  fused[g] = (__bf16)(fg * a + (1.f - fg) * b);
}

// ---------------- depthwise causal conv (DC=4) + SiLU ----------------
__global__ void k_conv(const float* __restrict__ xz, const float* __restrict__ cw,
                       const float* __restrict__ cb, float* __restrict__ xc) {
  long g = blockIdx.x * 256L + threadIdx.x;
  if (g >= (long)NN * DIi) return;
  int i = (int)(g & 255); long n = g >> 8;
  float acc = cb[i];
#pragma unroll
  for (int k = 0; k < 4; ++k) {
    long mrow = n - 3 + k;
    if (mrow >= 0) acc += xz[(size_t)mrow * 512 + i] * cw[i * 4 + k];
  }
  xc[g] = acc / (1.f + __expf(-acc));
}

__global__ void k_xdbl(const float* __restrict__ xc, const float* __restrict__ Wxp,
                       float* __restrict__ xdbl) {
  long g = blockIdx.x * 256L + threadIdx.x;       // N*40
  if (g >= (long)NN * 40) return;
  int j = (int)(g % 40); long n = g / 40;
  const float* xr = xc + n * DIi;
  const float* wr = Wxp + (size_t)j * DIi;
  float acc = 0.f;
#pragma unroll 4
  for (int i = 0; i < DIi; ++i) acc += xr[i] * wr[i];
  xdbl[g] = acc;
}

__global__ void k_delta(const float* __restrict__ xdbl, const float* __restrict__ Wdt,
                        const float* __restrict__ bdt, float* __restrict__ delta) {
  long g = blockIdx.x * 256L + threadIdx.x;       // N*256
  if (g >= (long)NN * DIi) return;
  int i = (int)(g & 255); long n = g >> 8;
  const float* dt = xdbl + n * 40;
  float acc = bdt[i];
#pragma unroll
  for (int t = 0; t < 8; ++t) acc += dt[t] * Wdt[i * 8 + t];
  delta[g] = log1pf(__expf(-fabsf(acc))) + fmaxf(acc, 0.f);   // softplus
}

// ---------------- chunked diagonal scan: pass A (local P,H with h0=0) ----------------
__global__ void k_scanA(const float* __restrict__ delta, const float* __restrict__ xdbl,
                        const float* __restrict__ xc, const float* __restrict__ Alog,
                        float* __restrict__ P, float* __restrict__ H) {
  long g = blockIdx.x * 256L + threadIdx.x;
  if (g >= (long)NCHUNK * 4096) return;
  int c = (int)(g >> 12); int q = (int)(g & 4095);
  int i = q >> 4, s = q & 15;
  float a = -__expf(Alog[i * DSs + s]);
  float h = 0.f, p = 1.f;
  int n0 = c * LCH, n1 = min(NN, n0 + LCH);
  for (int n = n0; n < n1; ++n) {
    __builtin_prefetch(delta + (size_t)(n + 4) * DIi + i, 0, 0);  // global_prefetch_b8
    __builtin_prefetch(xc    + (size_t)(n + 4) * DIi + i, 0, 0);
    float d  = delta[(size_t)n * DIi + i];
    float da = __expf(d * a);
    h = da * h + d * xdbl[(size_t)n * 40 + 8 + s] * xc[(size_t)n * DIi + i];
    p *= da;
  }
  P[g] = p; H[g] = h;
}

// pass B: serial over 157 chunks, 4096 independent lanes
__global__ void k_scanB(const float* __restrict__ P, const float* __restrict__ H,
                        float* __restrict__ S) {
  int q = blockIdx.x * 256 + threadIdx.x;
  if (q >= 4096) return;
  float h = 0.f;
  for (int c = 0; c < NCHUNK; ++c) {
    S[(size_t)c * 4096 + q] = h;
    h = P[(size_t)c * 4096 + q] * h + H[(size_t)c * 4096 + q];
  }
}

// pass C: replay with correct h0, reduce over DS=16 lanes, fuse skip+SiLU(z)
__global__ void k_scanC(const float* __restrict__ delta, const float* __restrict__ xdbl,
                        const float* __restrict__ xc, const float* __restrict__ xz,
                        const float* __restrict__ Alog, const float* __restrict__ Dsk,
                        const float* __restrict__ S, float* __restrict__ y) {
  long g = blockIdx.x * 256L + threadIdx.x;
  if (g >= (long)NCHUNK * 4096) return;
  int c = (int)(g >> 12); int q = (int)(g & 4095);
  int i = q >> 4, s = q & 15;
  float a = -__expf(Alog[i * DSs + s]);
  float h = S[g];
  float dsk = Dsk[i];
  int n0 = c * LCH, n1 = min(NN, n0 + LCH);
  for (int n = n0; n < n1; ++n) {
    __builtin_prefetch(delta + (size_t)(n + 4) * DIi + i, 0, 0);
    __builtin_prefetch(xc    + (size_t)(n + 4) * DIi + i, 0, 0);
    float xcv = xc[(size_t)n * DIi + i];
    float d   = delta[(size_t)n * DIi + i];
    float da  = __expf(d * a);
    h = da * h + d * xdbl[(size_t)n * 40 + 8 + s] * xcv;
    float part = h * xdbl[(size_t)n * 40 + 24 + s];
    part += __shfl_xor(part, 1); part += __shfl_xor(part, 2);   // xor<16 stays inside the
    part += __shfl_xor(part, 4); part += __shfl_xor(part, 8);   // 16-lane state group
    if (s == 0) {
      float zc = xz[(size_t)n * 512 + DIi + i];
      y[(size_t)n * DIi + i] = (part + dsk * xcv) * (zc / (1.f + __expf(-zc)));
    }
  }
}

// ---------------- final residual + LN + BN-affine (1 wave / row) ----------------
__global__ void k_final(const float* __restrict__ mt, const float* __restrict__ xin,
                        const float* __restrict__ lw, const float* __restrict__ lb,
                        const float* __restrict__ bw, const float* __restrict__ bb,
                        const float* __restrict__ brm, const float* __restrict__ brv,
                        float* __restrict__ out) {
  int row = blockIdx.x * 8 + (threadIdx.x >> 5);
  if (row >= NN) return;
  int lane = threadIdx.x & 31;
  float v[4], s = 0.f, s2 = 0.f;
#pragma unroll
  for (int t = 0; t < 4; ++t) {
    int j = lane * 4 + t;
    float x = mt[(size_t)row * DD + j] + xin[(size_t)row * DD + j];
    v[t] = x; s += x; s2 += x * x;
  }
  s = wave_red32(s); s2 = wave_red32(s2);
  float m = s * (1.f / DD);
  float inv = rsqrtf(s2 * (1.f / DD) - m * m + 1e-5f);
#pragma unroll
  for (int t = 0; t < 4; ++t) {
    int j = lane * 4 + t;
    float o = (v[t] - m) * inv * lw[j] + lb[j];
    out[(size_t)row * DD + j] = (o - brm[j]) * rsqrtf(brv[j] + 1e-5f) * bw[j] + bb[j];
  }
}

// =======================================================================================
static inline size_t alup(size_t x) { return (x + 255) & ~(size_t)255; }

extern "C" void kernel_launch(void* const* d_in, const int* in_sizes, int n_in,
                              void* d_out, int out_size, void* d_ws, size_t ws_size,
                              hipStream_t stream) {
  const float* x    = (const float*)d_in[0];
  const int*   ei   = (const int*)  d_in[1];
  const float* ew   = (const float*)d_in[2];
  const float* gu   = (const float*)d_in[3];
  const float* W1   = (const float*)d_in[4];
  const float* b1   = (const float*)d_in[5];
  const float* lnw  = (const float*)d_in[6];
  const float* lnb  = (const float*)d_in[7];
  const float* Wg   = (const float*)d_in[8];
  const float* bg   = (const float*)d_in[9];
  const float* Wf   = (const float*)d_in[10];
  const float* bfv  = (const float*)d_in[11];
  const float* Wv1  = (const float*)d_in[12];
  const float* bv1  = (const float*)d_in[13];
  const float* Wv2  = (const float*)d_in[14];
  const float* bv2  = (const float*)d_in[15];
  const float* Win  = (const float*)d_in[16];
  const float* cw   = (const float*)d_in[17];
  const float* cb   = (const float*)d_in[18];
  const float* Wxp  = (const float*)d_in[19];
  const float* Wdt  = (const float*)d_in[20];
  const float* bdt  = (const float*)d_in[21];
  const float* Alog = (const float*)d_in[22];
  const float* Dsk  = (const float*)d_in[23];
  const float* Wout = (const float*)d_in[24];
  const float* low  = (const float*)d_in[25];
  const float* lob  = (const float*)d_in[26];
  const float* bnw  = (const float*)d_in[27];
  const float* bnb  = (const float*)d_in[28];
  const float* bnrm = (const float*)d_in[29];
  const float* bnrv = (const float*)d_in[30];

  float* out = (float*)d_out;
  float* xin = out + (size_t)NN * DD;              // second tuple output: x_in

  // ---- workspace layout (lifetime-based region reuse) ----
  char* ws = (char*)d_ws;
  size_t off = 0;
  size_t oR1 = off;  off += alup((size_t)NN * FIN * 2);        // xbf  -> later y (N*256*4, same size)
  size_t oR2 = off;  off += alup((size_t)NN * DD * 4 * 2);     // x_norm+view2 -> later delta
  size_t oR3 = off;  off += alup((size_t)NN * 2 * DD * 2);     // hcat -> later ybf
  size_t oG  = off;  off += alup((size_t)NN * 4);              // gate
  size_t oW1t = off; off += alup((size_t)DD * FIN * 2);
  size_t oWc  = off; off += alup((size_t)384 * 256 * 2);
  size_t oWin = off; off += alup((size_t)512 * 128 * 2);
  size_t oWo  = off; off += alup((size_t)128 * 256 * 2);
  size_t oR6 = off;  off += alup((size_t)NN * 384 * 4);        // tmp384 -> later xc + mamba_tmp
  size_t oFu = off;  off += alup((size_t)NN * DD * 2);         // fused bf16
  size_t oXz = off;  off += alup((size_t)NN * 512 * 4);        // xz
  size_t oXd = off;  off += alup((size_t)NN * 40 * 4);         // x_dbl
  size_t oP  = off;  off += alup((size_t)NCHUNK * 4096 * 4);
  size_t oH  = off;  off += alup((size_t)NCHUNK * 4096 * 4);
  size_t oS  = off;  off += alup((size_t)NCHUNK * 4096 * 4);
  (void)ws_size; (void)in_sizes; (void)n_in; (void)out_size;

  __bf16* xbf    = (__bf16*)(ws + oR1);
  float*  yv     = (float*) (ws + oR1);
  float*  xnorm  = (float*) (ws + oR2);
  float*  view2  = (float*) (ws + oR2) + (size_t)NN * DD;
  float*  delta  = (float*) (ws + oR2);
  __bf16* hcat   = (__bf16*)(ws + oR3);
  __bf16* ybf    = (__bf16*)(ws + oR3);
  float*  gate   = (float*) (ws + oG);
  __bf16* W1t    = (__bf16*)(ws + oW1t);
  __bf16* Wcat   = (__bf16*)(ws + oWc);
  __bf16* WinT   = (__bf16*)(ws + oWin);
  __bf16* WoT    = (__bf16*)(ws + oWo);
  float*  tmp384 = (float*) (ws + oR6);
  float*  xc     = (float*) (ws + oR6);
  float*  mtmp   = (float*) (ws + oR6) + (size_t)NN * DIi;
  __bf16* fused  = (__bf16*)(ws + oFu);
  float*  xz     = (float*) (ws + oXz);
  float*  xdbl   = (float*) (ws + oXd);
  float*  scP    = (float*) (ws + oP);
  float*  scH    = (float*) (ws + oH);
  float*  scS    = (float*) (ws + oS);

  // ---- weight prep (bf16, B pre-transposed to [Ncols][K]) ----
  k_cvt_bf16<<<(unsigned)(((long)NN * FIN + 255) / 256), 256, 0, stream>>>(x, xbf, (long)NN * FIN);
  k_w1t <<<(DD * FIN) / 256, 256, 0, stream>>>(W1, W1t);
  k_wcat<<<(384 * 256) / 256, 256, 0, stream>>>(Wf, Wv1, Wv2, Wcat);
  k_cvt_bf16<<<(512 * 128) / 256, 256, 0, stream>>>(Win, WinT, 512 * 128);   // W_in already [j][k]
  k_cvt_bf16<<<(128 * 256) / 256, 256, 0, stream>>>(Wout, WoT, 128 * 256);   // W_out already [j][i]

  // ---- GEMM1: x_in = x @ W1 + b1 ----
  { int tiles = (NN / 16) * (DD / 64);
    k_gemm_bf16<<<(tiles + 3) / 4, 128, 0, stream>>>(xbf, W1t, xin, b1, NN, FIN, DD); }

  // ---- LN + gate; zero view2; gated scatter; hcat[:,128:] ----
  k_ln_gate<<<(NN + 7) / 8, 256, 0, stream>>>(xin, lnw, lnb, Wg, bg, gu, xnorm, hcat, gate);
  hipMemsetAsync(view2, 0, (size_t)NN * DD * 4, stream);
  k_scatter<<<(EE + 7) / 8, 256, 0, stream>>>(ei, ew, gate, xnorm, view2);
  k_v2hcat<<<(unsigned)(((long)NN * DD + 255) / 256), 256, 0, stream>>>(view2, hcat);

  // ---- fusion: one 256->384 GEMM + elementwise gate ----
  { int tiles = (NN / 16) * (384 / 64);
    k_gemm_bf16<<<(tiles + 3) / 4, 128, 0, stream>>>(hcat, Wcat, tmp384, nullptr, NN, 2 * DD, 384); }
  k_fused<<<(unsigned)(((long)NN * DD + 255) / 256), 256, 0, stream>>>(tmp384, bfv, bv1, bv2, fused);

  // ---- xz = fused @ W_in^T ----
  { int tiles = (NN / 16) * (512 / 64);
    k_gemm_bf16<<<(tiles + 3) / 4, 128, 0, stream>>>(fused, WinT, xz, nullptr, NN, DD, 512); }

  // ---- conv+SiLU, x_dbl, delta ----
  k_conv <<<(unsigned)(((long)NN * DIi + 255) / 256), 256, 0, stream>>>(xz, cw, cb, xc);
  k_xdbl <<<(unsigned)(((long)NN * 40 + 255) / 256), 256, 0, stream>>>(xc, Wxp, xdbl);
  k_delta<<<(unsigned)(((long)NN * DIi + 255) / 256), 256, 0, stream>>>(xdbl, Wdt, bdt, delta);

  // ---- 3-pass chunked diagonal scan ----
  k_scanA<<<NCHUNK * 16, 256, 0, stream>>>(delta, xdbl, xc, Alog, scP, scH);
  k_scanB<<<16, 256, 0, stream>>>(scP, scH, scS);
  k_scanC<<<NCHUNK * 16, 256, 0, stream>>>(delta, xdbl, xc, xz, Alog, Dsk, scS, yv);

  // ---- mamba_out = y @ W_out^T ; final LN+BN ----
  k_cvt_bf16<<<(unsigned)(((long)NN * DIi + 255) / 256), 256, 0, stream>>>(yv, ybf, (long)NN * DIi);
  { int tiles = (NN / 16) * (DD / 64);
    k_gemm_bf16<<<(tiles + 3) / 4, 128, 0, stream>>>(ybf, WoT, mtmp, nullptr, NN, DIi, DD); }
  k_final<<<(NN + 7) / 8, 256, 0, stream>>>(mtmp, xin, low, lob, bnw, bnb, bnrm, bnrv, out);
}
